// DeltaNet_73907797229665
// MI455X (gfx1250) — matrix operand order repaired and checked by reading, not compiled, not run on hardware
//
#include <hip/hip_runtime.h>
#include <hip/hip_bf16.h>
#include <math.h>

// ---------------------------------------------------------------------------
// Problem constants (match reference)
// ---------------------------------------------------------------------------
#define BB 8
#define LL 2048
#define DD 1024
#define HH 4
#define HDIM 256
#define KTAPS 4

typedef __attribute__((ext_vector_type(16))) __bf16 v16bf;
typedef __attribute__((ext_vector_type(8)))  float  v8f;
typedef __attribute__((ext_vector_type(4)))  float  f4;
typedef __attribute__((ext_vector_type(4)))  unsigned int u4;

union BF16x16 { v16bf v; __bf16 e[16]; unsigned int u[8]; };
union V8F     { v8f   v; float  e[8];  };

__device__ inline float wave_reduce_sum(float v) {
#pragma unroll
  for (int off = 16; off > 0; off >>= 1) v += __shfl_xor(v, off, 32);
  return v;
}

__device__ inline float silu_f(float x) { return x / (1.0f + __expf(-x)); }

// ---------------------------------------------------------------------------
// Kernel 1: Y[m,n] = sum_k X[m,k] * W[n,k]   (bf16 WMMA, f32 accumulate)
// Block tile = 128(M) x 64(N), 8 waves, each wave 16x64.
// B tile staged cooperatively in LDS *as bf16* (conversion amortized 8x),
// double buffered, one barrier per 32-wide K step.
// LDS B layout: col c (0..63) -> 20 dwords (32 bf16 K-values, padded 16->20
// dwords so the 16 fragment lanes hit 16 distinct banks, 16B aligned).
// Grid must tile M,N exactly (no early exit: kernel contains barriers).
// ---------------------------------------------------------------------------
__global__ __launch_bounds__(256) void gemm_bf16_wmma_kernel(
    const float* __restrict__ X, const float* __restrict__ W,
    float* __restrict__ Y, int M, int N, int Kd) {
  __shared__ unsigned int bsh[2][64 * 20];

  const int tid  = threadIdx.x;
  const int lane = tid & 31;
  const int wid  = tid >> 5;
  const int tilesN = N >> 6;
  const int blockM = blockIdx.x / tilesN;
  const int blockN = blockIdx.x % tilesN;
  const int row0 = blockM * 128 + wid * 16;
  const int col0 = blockN * 64;

  const int m  = lane & 15;
  const int hi = lane >> 4;

  v8f acc[4] = {v8f{}, v8f{}, v8f{}, v8f{}};

  // A: per-lane row pointer (frag: e[0..7]=K+base.., e[8..15]=K+base+16..,
  // base = hi?8:0)
  const float* arow = X + (size_t)(row0 + m) * Kd + (hi ? 8 : 0);

  // B staging: 256 threads cover 64 cols x 4 parts (8 K-floats each)
  const int sc = tid >> 2;        // column 0..63
  const int sp = tid & 3;         // K-part 0..3 (8 floats -> 4 dwords bf16)
  const float* sgp = W + (size_t)(col0 + sc) * Kd + sp * 8;
  unsigned int* sdst0 = &bsh[0][(size_t)sc * 20 + sp * 4];
  unsigned int* sdst1 = &bsh[1][(size_t)sc * 20 + sp * 4];

  auto stage = [&](int buf, int k0) {
    f4 x0 = *(const f4*)(sgp + k0);
    f4 x1 = *(const f4*)(sgp + k0 + 4);
    union { __bf16 h[8]; u4 v; } t;
#pragma unroll
    for (int e = 0; e < 4; ++e) {
      t.h[e]     = (__bf16)x0[e];
      t.h[4 + e] = (__bf16)x1[e];
    }
    *(u4*)(buf ? sdst1 : sdst0) = t.v;
  };

  stage(0, 0);
  __syncthreads();

  int buf = 0;
  for (int k0 = 0; k0 < Kd; k0 += 32, buf ^= 1) {
    if (k0 + 32 < Kd) {
      __builtin_prefetch(sgp + k0 + 64, 0, 3);   // near-cache prefetch
      stage(buf ^ 1, k0 + 32);
    }

    // ---- A fragment (4 x b128 global, one clause) ----
    BF16x16 af;
    {
      const float* p = arow + k0;
      f4 a0 = *(const f4*)(p);
      f4 a1 = *(const f4*)(p + 4);
      f4 a2 = *(const f4*)(p + 16);
      f4 a3 = *(const f4*)(p + 20);
#pragma unroll
      for (int e = 0; e < 4; ++e) {
        af.e[e]      = (__bf16)a0[e];
        af.e[4 + e]  = (__bf16)a1[e];
        af.e[8 + e]  = (__bf16)a2[e];
        af.e[12 + e] = (__bf16)a3[e];
      }
    }

    // ---- 4 B fragments from LDS (2 x ds_load_b128 each) + WMMA ----
    BF16x16 bf[4];
#pragma unroll
    for (int cj = 0; cj < 4; ++cj) {
      const unsigned int* pc =
          &bsh[buf][(size_t)(cj * 16 + m) * 20 + hi * 4];
      u4 lo = *(const u4*)(pc);
      u4 hw = *(const u4*)(pc + 8);
#pragma unroll
      for (int e = 0; e < 4; ++e) { bf[cj].u[e] = lo[e]; bf[cj].u[4 + e] = hw[e]; }
    }
#pragma unroll
    for (int cj = 0; cj < 4; ++cj)
      acc[cj] = __builtin_amdgcn_wmma_f32_16x16x32_bf16(
          false, af.v, false, bf[cj].v, (short)0, acc[cj], false, false);

    // Scheduling hint: cluster the 8 B-frag DS reads, then the 4 WMMAs,
    // instead of ds_load/wait/wmma serialization (inert if unsupported).
    __builtin_amdgcn_sched_group_barrier(0x100, 8, 0);  // DS reads
    __builtin_amdgcn_sched_group_barrier(0x008, 4, 0);  // MFMA/WMMA

    __syncthreads();
  }

  // ---- store (C/D layout: VGPR r -> M=r(+8 if hi), N=lane&15) ----
  const int mo = hi * 8;
  V8F u0, u1, u2, u3;
  u0.v = acc[0]; u1.v = acc[1]; u2.v = acc[2]; u3.v = acc[3];
#pragma unroll
  for (int r = 0; r < 8; ++r) {
    size_t off = (size_t)(row0 + mo + r) * N + col0 + m;
    Y[off]      = u0.e[r];
    Y[off + 16] = u1.e[r];
    Y[off + 32] = u2.e[r];
    Y[off + 48] = u3.e[r];
  }
}

// ---------------------------------------------------------------------------
// Kernel 2: in-place causal depthwise conv (K=4) + double SiLU.
// One thread per (b,d) channel, rolling register window.
// ---------------------------------------------------------------------------
__global__ void conv_silu2_kernel(float* __restrict__ Y,
                                  const float* __restrict__ Wc) {
  int idx = blockIdx.x * blockDim.x + threadIdx.x;
  if (idx >= BB * DD) return;
  int b = idx / DD;
  int d = idx % DD;
  float w0 = Wc[d * KTAPS + 0], w1 = Wc[d * KTAPS + 1];
  float w2 = Wc[d * KTAPS + 2], w3 = Wc[d * KTAPS + 3];
  float h0 = 0.0f, h1 = 0.0f, h2 = 0.0f;
  float* base = Y + (size_t)b * LL * DD + d;
  for (int l = 0; l < LL; ++l) {
    float cur = base[(size_t)l * DD];
    float c   = h0 * w0 + h1 * w1 + h2 * w2 + cur * w3;
    base[(size_t)l * DD] = silu_f(silu_f(c));
    h0 = h1; h1 = h2; h2 = cur;
  }
}

// ---------------------------------------------------------------------------
// Kernel 3: in-place L2 normalization over head_dim=256 (wave per group).
// ---------------------------------------------------------------------------
__global__ void l2norm_kernel(float* __restrict__ Y, int ngroups) {
  int gw = blockIdx.x * (blockDim.x >> 5) + (threadIdx.x >> 5);
  if (gw >= ngroups) return;
  int lane = threadIdx.x & 31;
  float* p = Y + (size_t)gw * HDIM;
  float vals[8];
  float ss = 0.0f;
#pragma unroll
  for (int e = 0; e < 8; ++e) {
    vals[e] = p[lane + e * 32];
    ss += vals[e] * vals[e];
  }
  ss = wave_reduce_sum(ss);
  float inv = 1.0f / fmaxf(sqrtf(ss), 1e-12f);
#pragma unroll
  for (int e = 0; e < 8; ++e) p[lane + e * 32] = vals[e] * inv;
}

// ---------------------------------------------------------------------------
// Kernel 4: beta[b,l,h] = sigmoid( x[b,l,:] . Wbeta[h,:] ), wave per (b,l).
// ---------------------------------------------------------------------------
__global__ void beta_kernel(const float* __restrict__ X,
                            const float* __restrict__ Wb,
                            float* __restrict__ Beta) {
  int gw = blockIdx.x * (blockDim.x >> 5) + (threadIdx.x >> 5);
  if (gw >= BB * LL) return;
  int lane = threadIdx.x & 31;
  const float* xr = X + (size_t)gw * DD;
  float acc[HH] = {0.f, 0.f, 0.f, 0.f};
  for (int k = lane; k < DD; k += 32) {
    float xv = xr[k];
#pragma unroll
    for (int h = 0; h < HH; ++h) acc[h] += xv * Wb[(size_t)h * DD + k];
  }
#pragma unroll
  for (int h = 0; h < HH; ++h) acc[h] = wave_reduce_sum(acc[h]);
  if (lane == 0) {
#pragma unroll
    for (int h = 0; h < HH; ++h)
      Beta[(size_t)gw * HH + h] = 1.0f / (1.0f + __expf(-acc[h]));
  }
}

// ---------------------------------------------------------------------------
// Kernel 5: q is only consumed at t=L-1, so only rows l in {L-4..L-1} of the
// q projection are needed (conv window). 32 rows x 1024 cols, wave/output.
// qbuf layout: [b*4 + j][d], j = l-(L-4).
// ---------------------------------------------------------------------------
__global__ void qrows_gemm_kernel(const float* __restrict__ X,
                                  const float* __restrict__ Wq,
                                  float* __restrict__ qbuf) {
  int gw = blockIdx.x * (blockDim.x >> 5) + (threadIdx.x >> 5);
  if (gw >= 32 * DD) return;
  int lane = threadIdx.x & 31;
  int row = gw / DD;           // b*4 + j
  int n   = gw % DD;
  int b = row >> 2, j = row & 3;
  const float* xr = X + ((size_t)b * LL + (LL - 4 + j)) * DD;
  const float* wr = Wq + (size_t)n * DD;
  float acc = 0.0f;
  for (int k = lane; k < DD; k += 32) acc += xr[k] * wr[k];
  acc = wave_reduce_sum(acc);
  if (lane == 0) qbuf[(size_t)row * DD + n] = acc;
}

// ---------------------------------------------------------------------------
// Kernel 6: delta-rule scan, VGPR-resident state, depth-2 load pipeline.
// Rows of S evolve independently -> split the 256 rows of each (b,h) across
// SCAN_SPLIT=8 workgroups (32 rows each). Thread layout: 256 threads =
// 32 rows x 8 sections; each thread holds 32 S elements in registers.
// Global loads for step t+2 are issued at iteration t (values for step t were
// loaded two iterations ago and are dumped into the double-buffered LDS k
// line before the single per-step barrier), so L2/HBM latency is off the
// 2048-step critical path.
// ---------------------------------------------------------------------------
#define SCAN_SPLIT 8
#define ROWS_PER_WG 32
__global__ __launch_bounds__(256) void scan_kernel(
    const float* __restrict__ Kk, const float* __restrict__ V,
    const float* __restrict__ Beta, const float* __restrict__ qbuf,
    const float* __restrict__ cq, float* __restrict__ o_last,
    float* __restrict__ out) {
  __shared__ float kvb[2][HDIM];
  __shared__ float qv[HDIM];
  __shared__ float red[256];

  const int blk  = blockIdx.x;
  const int bh   = blk / SCAN_SPLIT;
  const int part = blk % SCAN_SPLIT;
  const int b = bh / HH, h = bh % HH;
  const int tid  = threadIdx.x;
  const int rloc = tid >> 3;                  // 0..31
  const int r    = part * ROWS_PER_WG + rloc; // state row 0..255
  const int j0   = (tid & 7) * 32;            // element section

  // ---- prologue: qv = l2norm(silu^2(conv(q_lin last rows))) ----
  {
    int d = h * HDIM + tid;
    float qc = 0.f;
#pragma unroll
    for (int j = 0; j < KTAPS; ++j)
      qc += qbuf[(size_t)(b * KTAPS + j) * DD + d] * cq[d * KTAPS + j];
    qc = silu_f(silu_f(qc));
    red[tid] = qc * qc;
    __syncthreads();
    for (int s = 128; s > 0; s >>= 1) {
      if (tid < s) red[tid] += red[tid + s];
      __syncthreads();
    }
    float inv = 1.0f / fmaxf(sqrtf(red[0]), 1e-12f);
    qv[tid] = qc * inv;
    __syncthreads();
  }

  float S[32];
#pragma unroll
  for (int j = 0; j < 32; ++j) S[j] = 0.f;

  // ---- depth-2 prefetch pipeline ----
  auto kbase = [&](int t) -> size_t {
    return ((size_t)b * LL + t) * DD + (size_t)h * HDIM;
  };
  auto bidx = [&](int t) -> size_t {
    return ((size_t)b * LL + t) * HH + h;
  };
  float kA = Kk[kbase(0) + tid], kB = Kk[kbase(1) + tid];
  float vA = V[kbase(0) + r],    vB = V[kbase(1) + r];
  float bA = Beta[bidx(0)],      bB = Beta[bidx(1)];

  for (int t = 0; t < LL; ++t) {
    kvb[t & 1][tid] = kA;
    int tf = (t + 2 < LL) ? (t + 2) : (LL - 1);
    float kC = Kk[kbase(tf) + tid];
    float vC = V[kbase(tf) + r];
    float bC = Beta[bidx(tf)];
    __syncthreads();

    const float* kv = kvb[t & 1] + j0;
    float sk = 0.f;
#pragma unroll
    for (int j = 0; j < 32; ++j) sk += S[j] * kv[j];
    sk += __shfl_xor(sk, 1, 32);
    sk += __shfl_xor(sk, 2, 32);
    sk += __shfl_xor(sk, 4, 32);
    float u = bA * (sk - vA);
#pragma unroll
    for (int j = 0; j < 32; ++j) S[j] -= u * kv[j];

    if (t == LL - 1) {
      const float* qp = qv + j0;
      float o = 0.f;
#pragma unroll
      for (int j = 0; j < 32; ++j) o += S[j] * qp[j];
      o += __shfl_xor(o, 1, 32);
      o += __shfl_xor(o, 2, 32);
      o += __shfl_xor(o, 4, 32);
      if ((tid & 7) == 0)
        o_last[(size_t)b * DD + (size_t)h * HDIM + r] = o;
    }

    kA = kB; kB = kC;
    vA = vB; vB = vC;
    bA = bB; bB = bC;
  }

  // write S_final: out layout = [o (B*D)] ++ [S (B,H,256,256)]
  float* So = out + (size_t)BB * DD + ((size_t)bh * HDIM + r) * HDIM + j0;
#pragma unroll
  for (int j = 0; j < 32; ++j) So[j] = S[j];
}

// ---------------------------------------------------------------------------
// Kernel 7a: per-batch RMS scale (8 waves, one block).
// ---------------------------------------------------------------------------
__global__ void oscale_kernel(const float* __restrict__ o_last,
                              float* __restrict__ oscale) {
  int b    = threadIdx.x >> 5;
  int lane = threadIdx.x & 31;
  if (b >= BB) return;
  float s = 0.f;
  for (int k = lane; k < DD; k += 32) {
    float t = o_last[(size_t)b * DD + k];
    s += t * t;
  }
  s = wave_reduce_sum(s);
  if (lane == 0) oscale[b] = rsqrtf(s / (float)DD + 1e-5f);
}

// ---------------------------------------------------------------------------
// Kernel 7b: out[b,n] = sum_k o[b,k]*scale[b]*rms_w[k] * Wout[n,k].
// One wave per output; all reads lane-strided (coalesced).
// ---------------------------------------------------------------------------
__global__ void out_gemv_kernel(const float* __restrict__ o_last,
                                const float* __restrict__ oscale,
                                const float* __restrict__ rms_w,
                                const float* __restrict__ Wout,
                                float* __restrict__ out) {
  int gw = blockIdx.x * (blockDim.x >> 5) + (threadIdx.x >> 5);
  if (gw >= BB * DD) return;
  int lane = threadIdx.x & 31;
  int b = gw / DD, n = gw % DD;
  const float* orow = o_last + (size_t)b * DD;
  const float* wr   = Wout + (size_t)n * DD;
  float acc = 0.f;
  for (int k = lane; k < DD; k += 32) acc += orow[k] * rms_w[k] * wr[k];
  acc = wave_reduce_sum(acc);
  if (lane == 0) out[(size_t)b * DD + n] = acc * oscale[b];
}

// ---------------------------------------------------------------------------
// Launch
// ---------------------------------------------------------------------------
extern "C" void kernel_launch(void* const* d_in, const int* in_sizes, int n_in,
                              void* d_out, int out_size, void* d_ws, size_t ws_size,
                              hipStream_t stream) {
  const float* x     = (const float*)d_in[0];
  const float* Wq    = (const float*)d_in[1];
  const float* Wk    = (const float*)d_in[2];
  const float* Wv    = (const float*)d_in[3];
  const float* cq    = (const float*)d_in[4];
  const float* ck    = (const float*)d_in[5];
  const float* cv    = (const float*)d_in[6];
  const float* Wbeta = (const float*)d_in[7];
  const float* rmsw  = (const float*)d_in[8];
  const float* Wout  = (const float*)d_in[9];
  float* out = (float*)d_out;

  // workspace: k,v (64MB each) + beta + o_last + qbuf + oscale (~128.3 MB)
  float* k      = (float*)d_ws;
  float* v      = k + (size_t)BB * LL * DD;
  float* beta   = v + (size_t)BB * LL * DD;
  float* o_last = beta + (size_t)BB * LL * HH;
  float* qbuf   = o_last + (size_t)BB * DD;
  float* oscale = qbuf + (size_t)32 * DD;

  const int M = BB * LL;  // 16384
  dim3 blk(256);

  // 1) K/V projections (WMMA bf16, LDS-staged B). Full q GEMM is dead work:
  //    q is only consumed at t = L-1 (conv window l = L-4..L-1).
  int gemmBlocks = (M / 128) * (DD / 64);  // 2048
  gemm_bf16_wmma_kernel<<<gemmBlocks, blk, 0, stream>>>(x, Wk, k, M, DD, DD);
  gemm_bf16_wmma_kernel<<<gemmBlocks, blk, 0, stream>>>(x, Wv, v, M, DD, DD);

  // 2) depthwise causal conv + double SiLU (in-place), k and v only
  int convBlocks = (BB * DD + 255) / 256;
  conv_silu2_kernel<<<convBlocks, blk, 0, stream>>>(k, ck);
  conv_silu2_kernel<<<convBlocks, blk, 0, stream>>>(v, cv);

  // 3) l2-normalize k per (b,l,h)
  int ngroups = BB * LL * HH;
  l2norm_kernel<<<(ngroups + 7) / 8, blk, 0, stream>>>(k, ngroups);

  // 4) beta gate
  beta_kernel<<<(BB * LL + 7) / 8, blk, 0, stream>>>(x, Wbeta, beta);

  // 5) q: only the last 4 positions per batch (32 rows x 1024)
  qrows_gemm_kernel<<<(32 * DD + 7) / 8, blk, 0, stream>>>(x, Wq, qbuf);

  // 6) delta-rule scan: 256 WGs (8 per (b,h)), VGPR-resident state
  scan_kernel<<<BB * HH * SCAN_SPLIT, blk, 0, stream>>>(k, v, beta, qbuf, cq,
                                                        o_last, out);

  // 7) RMSNorm scale + output projection
  oscale_kernel<<<1, blk, 0, stream>>>(o_last, oscale);
  out_gemv_kernel<<<(BB * DD + 7) / 8, blk, 0, stream>>>(o_last, oscale, rmsw,
                                                         Wout, out);
}
